// SimpleSSM_47476568490194
// MI455X (gfx1250) — compile-verified, matrix operand order checked by reading, not run on hardware
//
#include <hip/hip_runtime.h>
#include <hip/hip_fp16.h>
#include <math.h>

#define NDIM   256
#define SEQ    1024
#define RROWS  (NDIM * SEQ)      // 262144 rows of width 256
#define LAYERS 4
#define CHUNK  32                // L: steps per chunk
#define NCHUNK 32                // G: number of chunks (CHUNK*NCHUNK == SEQ)
#define LN_EPS 1e-5f

typedef __attribute__((ext_vector_type(16))) _Float16 v16h;
typedef __attribute__((ext_vector_type(4)))  _Float16 h4;
typedef __attribute__((ext_vector_type(8)))  float    v8f;
typedef __attribute__((ext_vector_type(2)))  float    v2f;
typedef __attribute__((ext_vector_type(4)))  unsigned int v4u;
typedef __attribute__((ext_vector_type(8)))  unsigned int v8u;

// LDS-resident W: row stride 264 halves (256 + 8 pad) -> ds_load_b128 lands on
// banks 4*lm..4*lm+3 per lane (conflict-free within a 16-lane half).
#define WSTRIDE 264
#define WLDS_BYTES (256 * WSTRIDE * 2)   // 135168
#define STAT_BYTES 2048                  // LN partial-stats area (512 floats)
#define TSTRIDE 258                      // padded f32 LDS row stride (fixup)

static __device__ inline v8f vzero8() {
  v8f z = {0.f,0.f,0.f,0.f,0.f,0.f,0.f,0.f};
  return z;
}

// f16 A/B fragment for 16x16x32: per lane, halves 0..7 <- K = kbase + 8*half + (0..7)
// halves 8..15 <- K = kbase + 16 + 8*half + (0..7). p = rowptr + kbase + 8*half.
static __device__ inline v16h ldfragh(const _Float16* p) {
  v16h r;
  reinterpret_cast<float4*>(&r)[0] = reinterpret_cast<const float4*>(p)[0];
  reinterpret_cast<float4*>(&r)[1] = reinterpret_cast<const float4*>(p + 16)[0];
  return r;
}

static __device__ inline v8f wmma_h(v16h a, v16h b, v8f c) {
  return __builtin_amdgcn_wmma_f32_16x16x32_f16(false, a, false, b, (short)0, c, false, false);
}
static __device__ inline v8f wmma_s(v2f a, v2f b, v8f c) {
  return __builtin_amdgcn_wmma_f32_16x16x4_f32(false, a, false, b, (short)0, c, false, false);
}

// ---------------------------------------------------------------------------
// Tensor Data Mover: stage a row-major 2D block into LDS offset 0 with padded
// row stride. rowE8 = row length in 8-byte elements; pad codes per D# spec.
// Issued once by wave 0; completion = s_wait_tensorcnt 0 + workgroup barrier.
static __device__ inline void tdm_stage(const void* gptr, unsigned rowE8,
                                        unsigned nrows, unsigned padI, unsigned padA) {
  if (threadIdx.x < 32) {
    unsigned long long ga = (unsigned long long)gptr;
    v4u g0;
    g0[0] = 1u;                                    // count = 1 valid descriptor
    g0[1] = 0u;                                    // lds_addr = 0
    g0[2] = (unsigned)(ga & 0xFFFFFFFFu);          // global_addr[31:0]
    g0[3] = (unsigned)((ga >> 32) & 0x01FFFFFFu)   // global_addr[56:32]
            | (2u << 30);                          // type = 2 (image)
    v8u g1;
    g1[0] = (3u << 16)        // data_size = 3 (8 bytes)
          | (1u << 20)        // pad_enable
          | (padI << 22)      // pad_interval: 2^(code+1) DWORDs
          | (padA << 25);     // pad_amount: code+1 DWORDs
    g1[1] = (rowE8 << 16);    // tensor_dim0 [bits 79:48]
    g1[2] = (nrows << 16);    // tensor_dim1 [bits 111:80]
    g1[3] = (rowE8 << 16);    // tile_dim0   [bits 127:112]
    g1[4] = nrows;            // tile_dim1   [bits 143:128]
    g1[5] = rowE8;            // tensor_dim0_stride [bits 207:160]
    g1[6] = 0u;
    g1[7] = 0u;
    asm volatile("tensor_load_to_lds %0, %1" :: "s"(g0), "s"(g1) : "memory");
    __builtin_amdgcn_s_wait_tensorcnt(0);
  }
  __syncthreads();
}

// ---------------------------------------------------------------------------
// f16 GEMM core, 8 col-tiles per wave: acc[8] (16 rows x 128 cols) =
// Xrow-panel @ W_lds^T slice. Linearized n = kb*8+j loop with a rotating
// 3-deep B-fragment pipeline and a double-buffered A fragment. With the 64-VGPR
// accumulator there is register headroom for the pipeline to materialize.
static __device__ inline void gemm_core_f16(const _Float16* xrow, const _Float16* WLb,
                                            v8f (&acc)[8]) {
  #pragma unroll
  for (int j = 0; j < 8; ++j) acc[j] = vzero8();
  v16h a_cur = ldfragh(xrow);
  v16h a_nxt = a_cur;
  v16h bb0 = ldfragh(WLb + (size_t)0 * 16 * WSTRIDE);
  v16h bb1 = ldfragh(WLb + (size_t)1 * 16 * WSTRIDE);
  v16h bb2 = ldfragh(WLb + (size_t)2 * 16 * WSTRIDE);
  #pragma unroll
  for (int n = 0; n < 64; ++n) {
    const int kb = n >> 3, j = n & 7;
    if (j == 0 && kb < 7) a_nxt = ldfragh(xrow + (kb + 1) * 32);
    v16h bc = (n % 3 == 0) ? bb0 : ((n % 3 == 1) ? bb1 : bb2);
    if (n + 3 < 64) {
      const int n3 = n + 3;
      v16h bn = ldfragh(WLb + (size_t)(n3 & 7) * 16 * WSTRIDE + (n3 >> 3) * 32);
      if (n % 3 == 0) bb0 = bn; else if (n % 3 == 1) bb1 = bn; else bb2 = bn;
    }
    acc[j] = wmma_h(a_cur, bc, acc[j]);
    if (j == 7) a_cur = a_nxt;
  }
}

// ---------------------------------------------------------------------------
// f32 256x256x256 core: acc[8] (one row-tile x 8 col-tiles per wave).
// A-fragments double-buffered from global; B-fragments depth-2 pipelined from
// LDS matrix `Mlds` with row stride `mstride`.
static __device__ inline void mm_core_f32(const float* arow, const float* Mlds,
                                          int mstride, int lm, int hf, int jB,
                                          v8f (&acc)[8]) {
  #pragma unroll
  for (int jj = 0; jj < 8; ++jj) acc[jj] = vzero8();
  v2f a = *reinterpret_cast<const v2f*>(arow);
  for (int kb = 0; kb < 64; ++kb) {
    v2f ac = a;
    if (kb < 63) a = *reinterpret_cast<const v2f*>(arow + (kb + 1) * 4);
    const int krow = kb * 4 + hf * 2;
    v2f b0, b1;
    b0.x = Mlds[krow * mstride + (jB + 0) * 16 + lm];
    b0.y = Mlds[(krow + 1) * mstride + (jB + 0) * 16 + lm];
    b1.x = Mlds[krow * mstride + (jB + 1) * 16 + lm];
    b1.y = Mlds[(krow + 1) * mstride + (jB + 1) * 16 + lm];
    #pragma unroll
    for (int jj = 0; jj < 8; ++jj) {
      v2f bc = (jj & 1) ? b1 : b0;
      if (jj + 2 < 8) {
        const int col = (jB + jj + 2) * 16 + lm;
        v2f bn;
        bn.x = Mlds[krow * mstride + col];
        bn.y = Mlds[(krow + 1) * mstride + col];
        if (jj & 1) b1 = bn; else b0 = bn;
      }
      acc[jj] = wmma_s(ac, bc, acc[jj]);
    }
  }
}

// ---------------------------------------------------------------- converts
__global__ void k_cvt_f16(const float4* __restrict__ s, h4* __restrict__ d, int n4) {
  for (int i = blockIdx.x * blockDim.x + threadIdx.x; i < n4; i += gridDim.x * blockDim.x) {
    float4 v = s[i];
    h4 o; o[0] = (_Float16)v.x; o[1] = (_Float16)v.y; o[2] = (_Float16)v.z; o[3] = (_Float16)v.w;
    d[i] = o;
  }
}

__global__ void k_copy_cvt(const float4* __restrict__ s, float4* __restrict__ d32,
                           h4* __restrict__ d16, int n4) {
  for (int i = blockIdx.x * blockDim.x + threadIdx.x; i < n4; i += gridDim.x * blockDim.x) {
    float4 v = s[i];
    d32[i] = v;
    h4 o; o[0] = (_Float16)v.x; o[1] = (_Float16)v.y; o[2] = (_Float16)v.z; o[3] = (_Float16)v.w;
    d16[i] = o;
  }
}

// --------------------------------------------------- phase 1: fused MLP GEMMs
// Row-panel GEMM: out[row, col] = sum_k X[row,k] * W[col,k]   (i.e. X @ W^T)
// Block = 512 threads = 16 waves; wave w owns row-tile (w>>1) and col-tile
// half (w&1)*8..+8, so a WG covers 128 rows x 256 cols with 64-VGPR accs.
// W comes from LDS (TDM-staged, 264-half padded stride): 2 WGs/WGP.

__global__ __launch_bounds__(512) void k_gemm_gelu(
    const _Float16* __restrict__ X, const _Float16* __restrict__ W,
    const float* __restrict__ bias, _Float16* __restrict__ H)
{
  extern __shared__ float ldsf[];
  const _Float16* WL = (const _Float16*)ldsf;
  tdm_stage(W, 64u, 256u, 6u, 3u);   // 512B rows + 16B pad -> 264-half stride

  const int tid = threadIdx.x;
  const int wv = tid >> 5, lane = tid & 31, hf = lane >> 4, lm = lane & 15;
  const int rt = wv >> 1, ch = wv & 1;
  const int rowbase = blockIdx.x * 128 + rt * 16;
  v8f acc[8];
  gemm_core_f16(X + (size_t)(rowbase + lm) * NDIM + hf * 8,
                WL + (size_t)(ch * 8 * 16 + lm) * WSTRIDE + hf * 8, acc);

  #pragma unroll
  for (int j = 0; j < 8; ++j) {
    const int col = (ch * 8 + j) * 16 + lm;
    const float bb = bias[col];
    #pragma unroll
    for (int r = 0; r < 8; ++r) {
      const int row = rowbase + r + 8 * hf;
      float v = acc[j][r] + bb;
      v = 0.5f * v * (1.0f + erff(v * 0.70710678118654752f));   // exact GELU
      H[(size_t)row * NDIM + col] = (_Float16)v;
    }
  }
}

__global__ __launch_bounds__(512) void k_gemm_ln_res(
    const _Float16* __restrict__ H, const _Float16* __restrict__ W,
    const float* __restrict__ bias, const float* __restrict__ gamma,
    const float* __restrict__ beta, float* __restrict__ Xbuf,
    _Float16* __restrict__ Xh)
{
  extern __shared__ float ldsf[];
  const _Float16* WL = (const _Float16*)ldsf;
  float* s1L = ldsf + WLDS_BYTES / 4;        // [128 rows][2 halves] partial sums
  float* s2L = s1L + 256;                    // [128 rows][2 halves] partial sumsq
  tdm_stage(W, 64u, 256u, 6u, 3u);

  const int tid = threadIdx.x;
  const int wv = tid >> 5, lane = tid & 31, hf = lane >> 4, lm = lane & 15;
  const int rt = wv >> 1, ch = wv & 1;
  const int rowbase = blockIdx.x * 128 + rt * 16;
  v8f acc[8];
  gemm_core_f16(H + (size_t)(rowbase + lm) * NDIM + hf * 8,
                WL + (size_t)(ch * 8 * 16 + lm) * WSTRIDE + hf * 8, acc);

  // bias + per-row partial stats over this wave's 128 columns
  float s1[8], s2[8];
  #pragma unroll
  for (int r = 0; r < 8; ++r) { s1[r] = 0.f; s2[r] = 0.f; }
  #pragma unroll
  for (int j = 0; j < 8; ++j) {
    const float bb = bias[(ch * 8 + j) * 16 + lm];
    #pragma unroll
    for (int r = 0; r < 8; ++r) {
      float v = acc[j][r] + bb;
      acc[j][r] = v;
      s1[r] += v;
      s2[r] += v * v;
    }
  }
  #pragma unroll
  for (int r = 0; r < 8; ++r) {
    #pragma unroll
    for (int m = 1; m < 16; m <<= 1) {
      s1[r] += __shfl_xor(s1[r], m, 16);
      s2[r] += __shfl_xor(s2[r], m, 16);
    }
  }
  // publish partials (one lane per 16-lane half), combine across the two
  // column-half waves of this row-tile
  if (lm == 0) {
    #pragma unroll
    for (int r = 0; r < 8; ++r) {
      const int rw = rt * 16 + r + 8 * hf;   // row index within WG [0,128)
      s1L[rw * 2 + ch] = s1[r];
      s2L[rw * 2 + ch] = s2[r];
    }
  }
  __syncthreads();
  float mu[8], rs[8];
  #pragma unroll
  for (int r = 0; r < 8; ++r) {
    const int rw = rt * 16 + r + 8 * hf;
    const float t1 = s1L[rw * 2 + 0] + s1L[rw * 2 + 1];
    const float t2 = s2L[rw * 2 + 0] + s2L[rw * 2 + 1];
    mu[r] = t1 * (1.0f / NDIM);
    rs[r] = rsqrtf(t2 * (1.0f / NDIM) - mu[r] * mu[r] + LN_EPS);
  }
  #pragma unroll
  for (int j = 0; j < 8; ++j) {
    const int col = (ch * 8 + j) * 16 + lm;
    const float g = gamma[col], bt = beta[col];
    #pragma unroll
    for (int r = 0; r < 8; ++r) {
      const int row = rowbase + r + 8 * hf;
      const size_t o = (size_t)row * NDIM + col;
      float y = (acc[j][r] - mu[r]) * rs[r] * g + bt + Xbuf[o];   // + residual
      Xbuf[o] = y;
      Xh[o] = (_Float16)y;
    }
  }
}

__global__ __launch_bounds__(512) void k_gemm_out32(
    const _Float16* __restrict__ X, const _Float16* __restrict__ W,
    float* __restrict__ O)
{
  extern __shared__ float ldsf[];
  const _Float16* WL = (const _Float16*)ldsf;
  tdm_stage(W, 64u, 256u, 6u, 3u);

  const int tid = threadIdx.x;
  const int wv = tid >> 5, lane = tid & 31, hf = lane >> 4, lm = lane & 15;
  const int rt = wv >> 1, ch = wv & 1;
  const int rowbase = blockIdx.x * 128 + rt * 16;
  v8f acc[8];
  gemm_core_f16(X + (size_t)(rowbase + lm) * NDIM + hf * 8,
                WL + (size_t)(ch * 8 * 16 + lm) * WSTRIDE + hf * 8, acc);

  #pragma unroll
  for (int j = 0; j < 8; ++j) {
    const int col = (ch * 8 + j) * 16 + lm;
    #pragma unroll
    for (int r = 0; r < 8; ++r) {
      const int row = rowbase + r + 8 * hf;
      O[(size_t)row * NDIM + col] = acc[j][r];
    }
  }
}

// -------------------------------------------- phase 2: chunked parallel scan
// All scan GEMMs are 256x256x256 in f32 WMMA (dynamic range faithful to ref:
// spectral radius of A is ~1.6 so the state grows and must track f32 overflow).
// 1024 threads = 32 waves; wave (w) owns row-tile i = w>>1, col-tiles (w&1)*8..+8.

// Apow[k] = A^k for k = 1..CHUNK  (slot 0 unused)
__global__ __launch_bounds__(1024) void k_powers(const float* __restrict__ A,
                                                 float* __restrict__ Apow)
{
  extern __shared__ float lds[];
  const int tid = threadIdx.x;
  const int wv = tid >> 5, lane = tid & 31, hf = lane >> 4, lm = lane & 15;
  const int iT = wv >> 1, jB = (wv & 1) * 8;

  for (int i = tid; i < NDIM * NDIM; i += 1024)
    lds[i] = ((i >> 8) == (i & 255)) ? 1.0f : 0.0f;   // identity
  __syncthreads();

  const float* arow = A + (size_t)(iT * 16 + lm) * NDIM + hf * 2;
  for (int k = 1; k <= CHUNK; ++k) {
    v8f acc[8];
    mm_core_f32(arow, lds, NDIM, lm, hf, jB, acc);
    __syncthreads();   // all reads of old M done
    float* dst = Apow + (size_t)k * (NDIM * NDIM);
    #pragma unroll
    for (int jj = 0; jj < 8; ++jj) {
      const int col = (jB + jj) * 16 + lm;
      #pragma unroll
      for (int r = 0; r < 8; ++r) {
        const int row = iT * 16 + r + 8 * hf;
        float v = acc[jj][r];
        lds[row * NDIM + col] = v;
        dst[(size_t)row * NDIM + col] = v;
      }
    }
    __syncthreads();
  }
}

// Local scans: chunk g, zero initial state:  Ml_t = A @ Ml_{t-1} + XB_t
__global__ __launch_bounds__(1024) void k_local_scan(const float* __restrict__ A,
                                                     const float* __restrict__ XB,
                                                     float* __restrict__ Ml)
{
  extern __shared__ float lds[];
  const int tid = threadIdx.x;
  const int wv = tid >> 5, lane = tid & 31, hf = lane >> 4, lm = lane & 15;
  const int iT = wv >> 1, jB = (wv & 1) * 8;
  const int g = blockIdx.x;

  for (int i = tid; i < NDIM * NDIM; i += 1024) lds[i] = 0.0f;
  __syncthreads();

  const float* arow = A + (size_t)(iT * 16 + lm) * NDIM + hf * 2;
  for (int j = 0; j < CHUNK; ++j) {
    const int t = g * CHUNK + j;
    v8f acc[8];
    mm_core_f32(arow, lds, NDIM, lm, hf, jB, acc);
    __syncthreads();
    #pragma unroll
    for (int jj = 0; jj < 8; ++jj) {
      const int col = (jB + jj) * 16 + lm;
      #pragma unroll
      for (int r = 0; r < 8; ++r) {
        const int brow = iT * 16 + r + 8 * hf;                    // batch index
        float v = acc[jj][r] + XB[((size_t)brow * SEQ + t) * NDIM + col];
        lds[brow * NDIM + col] = v;
        Ml[(size_t)t * (NDIM * NDIM) + brow * NDIM + col] = v;
      }
    }
    __syncthreads();
  }
}

// Serial carries: C_0 = 0; C_g = P @ C_{g-1} + Ml[g*CHUNK - 1]   (P = A^CHUNK)
__global__ __launch_bounds__(1024) void k_carry(const float* __restrict__ P,
                                                const float* __restrict__ Ml,
                                                float* __restrict__ Cb)
{
  extern __shared__ float lds[];
  const int tid = threadIdx.x;
  const int wv = tid >> 5, lane = tid & 31, hf = lane >> 4, lm = lane & 15;
  const int iT = wv >> 1, jB = (wv & 1) * 8;

  for (int i = tid; i < NDIM * NDIM; i += 1024) { lds[i] = 0.0f; Cb[i] = 0.0f; }
  __syncthreads();

  const float* prow = P + (size_t)(iT * 16 + lm) * NDIM + hf * 2;
  for (int g = 1; g < NCHUNK; ++g) {
    v8f acc[8];
    mm_core_f32(prow, lds, NDIM, lm, hf, jB, acc);
    __syncthreads();
    const float* S = Ml + (size_t)(g * CHUNK - 1) * (NDIM * NDIM);
    float* dst = Cb + (size_t)g * (NDIM * NDIM);
    #pragma unroll
    for (int jj = 0; jj < 8; ++jj) {
      const int col = (jB + jj) * 16 + lm;
      #pragma unroll
      for (int r = 0; r < 8; ++r) {
        const int row = iT * 16 + r + 8 * hf;
        float v = acc[jj][r] + S[(size_t)row * NDIM + col];
        lds[row * NDIM + col] = v;
        dst[(size_t)row * NDIM + col] = v;
      }
    }
    __syncthreads();
  }
}

// Fixup + output: one block per t.
//   TDM-stage C_g into padded LDS, then T = Ml_t + A^{j+1} @ C_g (LDS),
//   then y_t = T @ C^T + XD_t  -> d_out
__global__ __launch_bounds__(1024) void k_fixup_out(
    const float* __restrict__ Ml, const float* __restrict__ Apow,
    const float* __restrict__ Cb, const float* __restrict__ Cw,
    const float* __restrict__ XD, float* __restrict__ out)
{
  extern __shared__ float lds[];
  const int tid = threadIdx.x;
  const int wv = tid >> 5, lane = tid & 31, hf = lane >> 4, lm = lane & 15;
  const int iT = wv >> 1, jB = (wv & 1) * 8;
  const int t = blockIdx.x;
  const int g = t >> 5, j = t & 31;

  // stage C_g (f32 256x256) into LDS: 1024B rows + 8B pad -> 258-float stride
  tdm_stage(Cb + (size_t)g * (NDIM * NDIM), 128u, 256u, 7u, 1u);

  // stage 1: T = A^{j+1} @ C_g + Ml_t
  const float* Aj = Apow + (size_t)(j + 1) * (NDIM * NDIM);
  v8f acc[8];
  mm_core_f32(Aj + (size_t)(iT * 16 + lm) * NDIM + hf * 2, lds, TSTRIDE, lm, hf, jB, acc);
  __syncthreads();   // all C_g reads done before overwriting with T

  const float* Mt = Ml + (size_t)t * (NDIM * NDIM);
  #pragma unroll
  for (int jj = 0; jj < 8; ++jj) {
    const int col = (jB + jj) * 16 + lm;
    #pragma unroll
    for (int r = 0; r < 8; ++r) {
      const int row = iT * 16 + r + 8 * hf;
      lds[row * TSTRIDE + col] = acc[jj][r] + Mt[(size_t)row * NDIM + col];
    }
  }
  __syncthreads();

  // stage 2: y = T @ C^T + XD_t ; out[b, t, :]
  v8f acc2[8];
  #pragma unroll
  for (int jj = 0; jj < 8; ++jj) acc2[jj] = vzero8();
  const float* trow = lds + (size_t)(iT * 16 + lm) * TSTRIDE + hf * 2;
  v2f a = *reinterpret_cast<const v2f*>(trow);
  for (int kb = 0; kb < 64; ++kb) {
    v2f ac = a;
    if (kb < 63) a = *reinterpret_cast<const v2f*>(trow + (kb + 1) * 4);
    // depth-2 pipelined global fragments of C^T (L2-resident)
    v2f b0 = *reinterpret_cast<const v2f*>(Cw + (size_t)((jB + 0) * 16 + lm) * NDIM + kb * 4 + hf * 2);
    v2f b1 = *reinterpret_cast<const v2f*>(Cw + (size_t)((jB + 1) * 16 + lm) * NDIM + kb * 4 + hf * 2);
    #pragma unroll
    for (int jj = 0; jj < 8; ++jj) {
      v2f bc = (jj & 1) ? b1 : b0;
      if (jj + 2 < 8) {
        v2f bn = *reinterpret_cast<const v2f*>(
            Cw + (size_t)((jB + jj + 2) * 16 + lm) * NDIM + kb * 4 + hf * 2);
        if (jj & 1) b1 = bn; else b0 = bn;
      }
      acc2[jj] = wmma_s(ac, bc, acc2[jj]);
    }
  }
  #pragma unroll
  for (int jj = 0; jj < 8; ++jj) {
    const int colo = (jB + jj) * 16 + lm;
    #pragma unroll
    for (int r = 0; r < 8; ++r) {
      const int brow = iT * 16 + r + 8 * hf;
      const size_t o = ((size_t)brow * SEQ + t) * NDIM + colo;
      out[o] = acc2[jj][r] + XD[o];
    }
  }
}

// ---------------------------------------------------------------- dispatcher
extern "C" void kernel_launch(void* const* d_in, const int* in_sizes, int n_in,
                              void* d_out, int out_size, void* d_ws, size_t ws_size,
                              hipStream_t stream) {
  (void)in_sizes; (void)n_in; (void)out_size; (void)ws_size;
  const float* x     = (const float*)d_in[0];
  const float* A     = (const float*)d_in[1];
  const float* B     = (const float*)d_in[2];
  const float* Cw    = (const float*)d_in[3];
  const float* D     = (const float*)d_in[4];
  const float* W1    = (const float*)d_in[5];
  const float* b1    = (const float*)d_in[6];
  const float* W2    = (const float*)d_in[7];
  const float* b2    = (const float*)d_in[8];
  const float* gamma = (const float*)d_in[9];
  const float* beta  = (const float*)d_in[10];
  float* out = (float*)d_out;

  char* wp = (char*)d_ws;
  auto alloc = [&](size_t bytes) -> char* {
    char* p = wp;
    wp += (bytes + 255) & ~(size_t)255;
    return p;
  };
  const size_t MAT = (size_t)NDIM * NDIM;          // 65536
  float*    Xbuf = (float*)   alloc((size_t)RROWS * NDIM * 4);
  _Float16* Xh   = (_Float16*)alloc((size_t)RROWS * NDIM * 2);
  _Float16* Hh   = (_Float16*)alloc((size_t)RROWS * NDIM * 2);
  float*    XB   = (float*)   alloc((size_t)RROWS * NDIM * 4);
  float*    XD   = (float*)   alloc((size_t)RROWS * NDIM * 4);
  float*    Mlb  = (float*)   alloc((size_t)SEQ * MAT * 4);
  float*    Apow = (float*)   alloc((size_t)(CHUNK + 1) * MAT * 4);
  float*    Cb   = (float*)   alloc((size_t)NCHUNK * MAT * 4);
  _Float16* W1h  = (_Float16*)alloc((size_t)LAYERS * MAT * 2);
  _Float16* W2h  = (_Float16*)alloc((size_t)LAYERS * MAT * 2);
  _Float16* Bh   = (_Float16*)alloc(MAT * 2);
  _Float16* Dh   = (_Float16*)alloc(MAT * 2);

  // weight / input conversions (vectorized; all sizes divisible by 4)
  k_cvt_f16<<<128, 256, 0, stream>>>((const float4*)W1, (h4*)W1h, LAYERS * (int)MAT / 4);
  k_cvt_f16<<<128, 256, 0, stream>>>((const float4*)W2, (h4*)W2h, LAYERS * (int)MAT / 4);
  k_cvt_f16<<<32, 256, 0, stream>>>((const float4*)B, (h4*)Bh, (int)MAT / 4);
  k_cvt_f16<<<32, 256, 0, stream>>>((const float4*)D, (h4*)Dh, (int)MAT / 4);
  k_copy_cvt<<<1024, 256, 0, stream>>>((const float4*)x, (float4*)Xbuf, (h4*)Xh,
                                       RROWS * NDIM / 4);

  // phase 1: residual MLP stack (TDM-staged weights in LDS)
  for (int l = 0; l < LAYERS; ++l) {
    k_gemm_gelu<<<RROWS / 128, 512, WLDS_BYTES, stream>>>(
        Xh, W1h + (size_t)l * MAT, b1 + l * NDIM, Hh);
    k_gemm_ln_res<<<RROWS / 128, 512, WLDS_BYTES + STAT_BYTES, stream>>>(
        Hh, W2h + (size_t)l * MAT, b2 + l * NDIM,
        gamma + l * NDIM, beta + l * NDIM, Xbuf, Xh);
  }

  // phase 2 inputs: XB = X @ B^T, XD = X @ D^T (all rows in parallel)
  k_gemm_out32<<<RROWS / 128, 512, WLDS_BYTES, stream>>>(Xh, Bh, XB);
  k_gemm_out32<<<RROWS / 128, 512, WLDS_BYTES, stream>>>(Xh, Dh, XD);

  // chunked parallel scan on the collapsed 256x256 mean-state
  k_powers<<<1, 1024, NDIM * NDIM * 4, stream>>>(A, Apow);
  k_local_scan<<<NCHUNK, 1024, NDIM * NDIM * 4, stream>>>(A, XB, Mlb);
  k_carry<<<1, 1024, NDIM * NDIM * 4, stream>>>(Apow + (size_t)CHUNK * MAT, Mlb, Cb);
  k_fixup_out<<<SEQ, 1024, NDIM * TSTRIDE * 4, stream>>>(Mlb, Apow, Cb, Cw, XD, out);
}
